// MambaResBlock_74895639708090
// MI455X (gfx1250) — matrix operand order repaired and checked by reading, not compiled, not run on hardware
//
#include <hip/hip_runtime.h>

#define DM 512
#define PJ 1024
#define NS 32
#define BQ 2
#define LQ 2048
#define MROWS (BQ*LQ) /* 4096 */

typedef __attribute__((ext_vector_type(16))) __bf16 v16bf;
typedef __attribute__((ext_vector_type(8)))  float  v8f;
typedef unsigned int u32x4 __attribute__((ext_vector_type(4)));
typedef int          i32x4 __attribute__((ext_vector_type(4)));
typedef int          i32x8 __attribute__((ext_vector_type(8)));

#ifndef __has_builtin
#define __has_builtin(x) 0
#endif
#if __has_builtin(__builtin_amdgcn_tensor_load_to_lds)
#define USE_TDM 1
#else
#define USE_TDM 0
#endif

__device__ __forceinline__ unsigned short f2bf(float f) {
  unsigned int u = __float_as_uint(f);
  u += 0x7FFFu + ((u >> 16) & 1u);
  return (unsigned short)(u >> 16);
}

union FragAB { v16bf v; unsigned int u32[8]; unsigned short u16[16]; };

#if USE_TDM
// 2D TDM load: `rows` rows of 32 bf16 (64B), global row stride = strideElems*2 bytes,
// packed contiguously into LDS at byte offset lds_byte. Issued per-wave; TENSORcnt.
__device__ __forceinline__ void tdm_load_2d(unsigned lds_byte, unsigned long long ga,
                                            unsigned rows, unsigned strideElems) {
  u32x4 g0;
  g0.x = 1u;                                            // count=1, user mode
  g0.y = lds_byte;                                      // lds_addr
  g0.z = (unsigned)(ga & 0xFFFFFFFFu);                  // global_addr[31:0]
  g0.w = ((unsigned)((ga >> 32) & 0x01FFFFFFu)) | 0x80000000u; // addr[56:32] | type=2
  i32x8 g1;
  const unsigned td0 = 0x40000000u, td1 = 0x40000000u;  // huge dims: no OOB clipping
  g1[0] = 0x00010000;                                   // data_size=1 (2 bytes), mask=0
  g1[1] = (int)(td0 << 16);                             // abar=0 | tensor_dim0[15:0]
  g1[2] = (int)((td0 >> 16) | (td1 << 16));             // td0[31:16] | td1[15:0]
  g1[3] = (int)((td1 >> 16) | (32u << 16));             // td1[31:16] | tile_dim0=32
  g1[4] = (int)(rows & 0xFFFFu);                        // tile_dim1=rows, tile_dim2=0
  g1[5] = (int)strideElems;                             // tensor_dim0_stride[31:0]
  g1[6] = 0;                                            // stride0[47:32] | stride1 lo
  g1[7] = 0;
  i32x4 z4 = {0, 0, 0, 0};
#if defined(__clang_major__) && __clang_major__ >= 23
  i32x8 z8 = {0, 0, 0, 0, 0, 0, 0, 0};
  __builtin_amdgcn_tensor_load_to_lds(g0, g1, z4, z4, z8, 0);
#else
  __builtin_amdgcn_tensor_load_to_lds(g0, g1, z4, z4, 0);
#endif
}
#endif

// ---------------- converters ----------------
__global__ void k_f2bf(const float* __restrict__ in, unsigned short* __restrict__ out, int n) {
  int i = blockIdx.x * blockDim.x + threadIdx.x;
  if (i < n) out[i] = f2bf(in[i]);
}

// transpose + convert: W[K][N] f32 -> Wt[N][K] bf16
__global__ void k_wt(const float* __restrict__ in, unsigned short* __restrict__ out,
                     int K, int N) {
  int i = blockIdx.x * blockDim.x + threadIdx.x;        // N*K
  int n = i / K, k = i - n * K;
  out[i] = f2bf(in[(size_t)k * N + n]);
}

// [WB|WC] -> transposed bf16 Wt[64][1024]
__global__ void k_pack_wbc_t(const float* __restrict__ WB, const float* __restrict__ WC,
                             unsigned short* __restrict__ out) {
  int i = blockIdx.x * blockDim.x + threadIdx.x;        // 64*1024
  int n = i >> 10, k = i & (PJ - 1);
  float v = (n < NS) ? WB[k * NS + n] : WC[k * NS + (n - NS)];
  out[i] = f2bf(v);
}

__global__ void k_pack_bbc(const float* __restrict__ bB, const float* __restrict__ bC,
                           float* __restrict__ out) {
  int c = threadIdx.x;                                  // 64 threads
  out[c] = (c < NS) ? bB[c] : bC[c - NS];
}

// ---------------- LayerNorm -> bf16 ----------------
__global__ void k_ln(const float* __restrict__ x, const float* __restrict__ g,
                     const float* __restrict__ b, unsigned short* __restrict__ h) {
  __shared__ float ss[256], sq[256];
  int r = blockIdx.x, t = threadIdx.x;
  const float* xr = x + (size_t)r * DM;
  float v0 = xr[t], v1 = xr[t + 256];
  ss[t] = v0 + v1; sq[t] = v0 * v0 + v1 * v1;
  __syncthreads();
  for (int s = 128; s > 0; s >>= 1) {
    if (t < s) { ss[t] += ss[t + s]; sq[t] += sq[t + s]; }
    __syncthreads();
  }
  float mu  = ss[0] * (1.0f / DM);
  float var = sq[0] * (1.0f / DM) - mu * mu;
  float rs  = rsqrtf(var + 1e-3f);
  unsigned short* hr = h + (size_t)r * DM;
  hr[t]       = f2bf((v0 - mu) * rs * g[t]       + b[t]);
  hr[t + 256] = f2bf((v1 - mu) * rs * g[t + 256] + b[t + 256]);
}

// ---------------- causal depthwise conv (K=4) + SiLU ----------------
__global__ void k_conv_silu(const float* __restrict__ x1raw, const float* __restrict__ ck,
                            const float* __restrict__ cb, float* __restrict__ x1f,
                            unsigned short* __restrict__ x1b) {
  int i = blockIdx.x * blockDim.x + threadIdx.x;        // MROWS*PJ
  int row = i >> 10;
  int c   = i & (PJ - 1);
  int l   = row & (LQ - 1);
  float acc = cb[c];
#pragma unroll
  for (int j = 0; j < 4; j++) {
    int lj = l - 3 + j;
    if (lj >= 0) acc += x1raw[(size_t)(row - 3 + j) * PJ + c] * ck[j * PJ + c];
  }
  float s = acc / (1.0f + __expf(-acc));
  x1f[i] = s;
  x1b[i] = f2bf(s);
}

// ---------------- WMMA bf16 GEMM, TDM double-buffered, fused epilogue ----------------
// C[M,N] = act(A[M,K] @ Wt[N,K]^T + bias[N]) (+ res[M,N])
// A row-major [M,K]; weights pre-transposed Wt[N,K] (K contiguous).
// Block tile 128x64, BK=32, 256 threads (8 waves, 4x2).
// Dynamic LDS: 2 buffers x (A 8KB + B 4KB) = 24576 bytes, A at +0, B at +8192.
template <int MODE, bool RES>
__global__ void k_gemm(const unsigned short* __restrict__ Ag,
                       const unsigned short* __restrict__ Bt,
                       const float* __restrict__ bias,
                       const float* __restrict__ res,
                       float* __restrict__ C,
                       int M, int N, int K) {
  extern __shared__ unsigned short smem[];              // no static LDS: base offset 0
  const int t = threadIdx.x;
  const int mBase = blockIdx.y * 128;
  const int nBase = blockIdx.x * 64;
  const int wave = t >> 5, lane = t & 31;
  const int wm = wave >> 1, wn = wave & 1;
  const int m16 = lane & 15, kh = lane >> 4;
  const int nkt = K >> 5;

  v8f zero = {};
  v8f acc[2][2];
#pragma unroll
  for (int a = 0; a < 2; a++)
#pragma unroll
    for (int b = 0; b < 2; b++) acc[a][b] = zero;

#if USE_TDM
  if (wave == 0) {
    tdm_load_2d(0u,    (unsigned long long)(Ag + (size_t)mBase * K), 128u, (unsigned)K);
    tdm_load_2d(8192u, (unsigned long long)(Bt + (size_t)nBase * K),  64u, (unsigned)K);
    __builtin_amdgcn_s_wait_tensorcnt(0);
  }
  __syncthreads();
#endif

  for (int kt = 0; kt < nkt; kt++) {
    const int k0 = kt << 5;
    const int buf = kt & 1;
#if USE_TDM
    if (wave == 0 && kt + 1 < nkt) {                    // prefetch next K-slice via TDM
      unsigned nb = (unsigned)((buf ^ 1) * 12288);
      tdm_load_2d(nb, (unsigned long long)(Ag + (size_t)mBase * K + k0 + 32), 128u, (unsigned)K);
      tdm_load_2d(nb + 8192u, (unsigned long long)(Bt + (size_t)nBase * K + k0 + 32), 64u, (unsigned)K);
    }
    const unsigned short* As = smem + buf * 6144;
    const unsigned short* Bs = smem + buf * 6144 + 4096;
#else
    const unsigned short* As = smem;
    const unsigned short* Bs = smem + 4096;
    if (kt > 0) __syncthreads();                        // protect overwrite
    // A tile 128x32: 2 x uint4 per thread
#pragma unroll
    for (int i = 0; i < 2; i++) {
      int lin = t + i * 256;
      int row = lin >> 2, seg = lin & 3;
      uint4 q = *(const uint4*)(Ag + (size_t)(mBase + row) * K + k0 + seg * 8);
      ((uint4*)smem)[row * 4 + seg] = q;
    }
    // B tile 64(n) x 32(k): one uint4 per thread, contiguous copy
    {
      int row = t >> 2, seg = t & 3;
      uint4 q = *(const uint4*)(Bt + (size_t)(nBase + row) * K + k0 + seg * 8);
      ((uint4*)(smem + 4096))[row * 4 + seg] = q;
    }
    if (k0 + 32 < K)
      __builtin_prefetch(Ag + (size_t)(mBase + (t >> 1)) * K + k0 + 32, 0, 1);
    __syncthreads();
#endif

    const unsigned int* As32 = (const unsigned int*)As;
    const unsigned int* Bs32 = (const unsigned int*)Bs;
    FragAB af[2], bfr[2];
#pragma unroll
    for (int mt = 0; mt < 2; mt++) {
      int m = wm * 32 + mt * 16 + m16;
#pragma unroll
      for (int j = 0; j < 8; j++) {
        int kp = (j < 4) ? (j + 4 * kh) : (8 + (j - 4) + 4 * kh);  // ISA A 16x32 layout
        af[mt].u32[j] = As32[m * 16 + kp];
      }
    }
#pragma unroll
    for (int nt = 0; nt < 2; nt++) {
      int n = wn * 32 + nt * 16 + m16;
#pragma unroll
      for (int j = 0; j < 8; j++)
        bfr[nt].u32[j] = Bs32[n * 16 + (j & 3) + 4 * kh + 8 * (j >> 2)];  // ISA B 32x16 layout
    }
#pragma unroll
    for (int mt = 0; mt < 2; mt++)
#pragma unroll
      for (int nt = 0; nt < 2; nt++)
        acc[mt][nt] = __builtin_amdgcn_wmma_f32_16x16x32_bf16(
            false, af[mt].v, false, bfr[nt].v, (short)0, acc[mt][nt], false, false);

#if USE_TDM
    if (kt + 1 < nkt) {
      if (wave == 0) __builtin_amdgcn_s_wait_tensorcnt(0);
      __syncthreads();                                  // flip buffers
    }
#endif
  }

  // epilogue: C/D layout -> VGPR r: lanes0-15 M=r, lanes16-31 M=8+r
#pragma unroll
  for (int mt = 0; mt < 2; mt++) {
    int row0 = mBase + wm * 32 + mt * 16 + kh * 8;
#pragma unroll
    for (int nt = 0; nt < 2; nt++) {
      int col = nBase + wn * 32 + nt * 16 + m16;
      float bcol = bias[col];
#pragma unroll
      for (int r = 0; r < 8; r++) {
        int row = row0 + r;
        float v = acc[mt][nt][r] + bcol;
        if (MODE == 1) v = v / (1.0f + __expf(-v));
        else if (MODE == 2) v = (v > 20.0f) ? v : log1pf(__expf(v));
        if (RES) v += res[(size_t)row * N + col];
        C[(size_t)row * N + col] = v;
      }
    }
  }
}

// ---------------- selective scan: one wave per (b,d), lane = state n ----------------
__global__ void k_scan(const float* __restrict__ Alog, const float* __restrict__ Dsk,
                       const float* __restrict__ delta, const float* __restrict__ u_,
                       const float* __restrict__ BCm, const float* __restrict__ x2,
                       unsigned short* __restrict__ yx2) {
  int gw   = blockIdx.x * 8 + (threadIdx.x >> 5);       // 2048 waves
  int lane = threadIdx.x & 31;                          // n
  int b = gw >> 10;
  int d = gw & (PJ - 1);
  float an  = -__expf(Alog[d * NS + lane]);             // A = -exp(A_log)
  float dsk = Dsk[d];
  float s = 0.0f;
  size_t rowbase = (size_t)b * LQ;
  for (int l = 0; l < LQ; l++) {
    size_t row = rowbase + l;
    float dt = delta[row * PJ + d];
    float uu = u_[row * PJ + d];
    float g  = __expf(dt * an);
    s = g * s + (dt * uu) * BCm[row * 64 + lane];       // Bm in cols 0..31
    float p = s * BCm[row * 64 + 32 + lane];            // Cm in cols 32..63
#pragma unroll
    for (int off = 16; off > 0; off >>= 1) p += __shfl_xor(p, off, 32);
    if (lane == 0) {
      float y = p + uu * dsk;
      yx2[row * PJ + d] = f2bf(y * x2[row * PJ + d]);
    }
  }
}

// ---------------- workspace layout (bytes) ----------------
#define OFF_H16    0u          /* 4096*512  bf16            */
#define OFF_W1T    4194304u    /* Wt1 [1024][512]  bf16     */
#define OFF_W2T    5242880u    /* Wt2 [1024][512]  bf16     */
#define OFF_WDT    6291456u    /* Wtd [1024][1024] bf16     */
#define OFF_W3T    8388608u    /* Wt3 [512][1024]  bf16     */
#define OFF_WBCT   9437184u    /* Wtbc [64][1024]  bf16     */
#define OFF_BBC    9568256u    /* 64 f32                    */
#define OFF_X1RAW  9568512u    /* 4096*1024 f32; reused for x2 after conv */
#define OFF_X1F    26345728u
#define OFF_X1B    43122944u   /* 4096*1024 bf16            */
#define OFF_DELTA  51511552u
#define OFF_BC     68288768u   /* 4096*64 f32               */
#define OFF_YX2B   69337344u   /* 4096*1024 bf16            */

#define GEMM_LDS 24576u

extern "C" void kernel_launch(void* const* d_in, const int* in_sizes, int n_in,
                              void* d_out, int out_size, void* d_ws, size_t ws_size,
                              hipStream_t stream) {
  const float* x     = (const float*)d_in[0];
  const float* ln_g  = (const float*)d_in[1];
  const float* ln_b  = (const float*)d_in[2];
  const float* W1    = (const float*)d_in[3];
  const float* b1    = (const float*)d_in[4];
  const float* W2    = (const float*)d_in[5];
  const float* b2    = (const float*)d_in[6];
  const float* convk = (const float*)d_in[7];
  const float* convb = (const float*)d_in[8];
  const float* A_log = (const float*)d_in[9];
  const float* D_sk  = (const float*)d_in[10];
  const float* WB    = (const float*)d_in[11];
  const float* WC    = (const float*)d_in[13];
  const float* Wd    = (const float*)d_in[15];
  const float* bd    = (const float*)d_in[16];
  const float* W3    = (const float*)d_in[17];
  const float* b3    = (const float*)d_in[18];
  float* out = (float*)d_out;

  char* ws = (char*)d_ws;
  unsigned short* h16   = (unsigned short*)(ws + OFF_H16);
  unsigned short* W1t   = (unsigned short*)(ws + OFF_W1T);
  unsigned short* W2t   = (unsigned short*)(ws + OFF_W2T);
  unsigned short* Wdt   = (unsigned short*)(ws + OFF_WDT);
  unsigned short* W3t   = (unsigned short*)(ws + OFF_W3T);
  unsigned short* WBCt  = (unsigned short*)(ws + OFF_WBCT);
  float*          bBC   = (float*)(ws + OFF_BBC);
  float*          x1raw = (float*)(ws + OFF_X1RAW);
  float*          x2    = (float*)(ws + OFF_X1RAW);    // alias: x1raw consumed by conv first
  float*          x1f   = (float*)(ws + OFF_X1F);
  unsigned short* x1b   = (unsigned short*)(ws + OFF_X1B);
  float*          delta = (float*)(ws + OFF_DELTA);
  float*          BC    = (float*)(ws + OFF_BC);
  unsigned short* yx2b  = (unsigned short*)(ws + OFF_YX2B);

  // weight transpose + bf16 conversion (one-time per call, ~2.7M elements)
  k_wt<<<(DM * PJ) / 256, 256, 0, stream>>>(W1, W1t, DM, PJ);   // Wt1[1024][512]
  k_wt<<<(DM * PJ) / 256, 256, 0, stream>>>(W2, W2t, DM, PJ);
  k_wt<<<(PJ * PJ) / 256, 256, 0, stream>>>(Wd, Wdt, PJ, PJ);
  k_wt<<<(PJ * DM) / 256, 256, 0, stream>>>(W3, W3t, PJ, DM);   // Wt3[512][1024]
  k_pack_wbc_t<<<(64 * PJ) / 256, 256, 0, stream>>>(WB, WC, WBCt);
  k_pack_bbc<<<1, 64, 0, stream>>>((const float*)d_in[12], (const float*)d_in[14], bBC);

  // LayerNorm
  k_ln<<<MROWS, 256, 0, stream>>>(x, ln_g, ln_b, h16);

  // x1_raw = h @ W1 + b1
  k_gemm<0, false><<<dim3(PJ / 64, MROWS / 128), 256, GEMM_LDS, stream>>>(
      h16, W1t, b1, nullptr, x1raw, MROWS, PJ, DM);

  // causal depthwise conv + SiLU
  k_conv_silu<<<(MROWS * PJ) / 256, 256, 0, stream>>>(x1raw, convk, convb, x1f, x1b);

  // x2 = silu(h @ W2 + b2)  (overwrites x1raw, already consumed)
  k_gemm<1, false><<<dim3(PJ / 64, MROWS / 128), 256, GEMM_LDS, stream>>>(
      h16, W2t, b2, nullptr, x2, MROWS, PJ, DM);

  // delta = softplus(x1 @ Wd + bd)
  k_gemm<2, false><<<dim3(PJ / 64, MROWS / 128), 256, GEMM_LDS, stream>>>(
      x1b, Wdt, bd, nullptr, delta, MROWS, PJ, PJ);

  // [Bm|Cm] = x1 @ [WB|WC] + [bB|bC]
  k_gemm<0, false><<<dim3(1, MROWS / 128), 256, GEMM_LDS, stream>>>(
      x1b, WBCt, bBC, nullptr, BC, MROWS, 64, PJ);

  // selective scan -> (y * x2) in bf16
  k_scan<<<(BQ * PJ) / 8, 256, 0, stream>>>(A_log, D_sk, delta, x1f, BC, x2, yx2b);

  // out = (y*x2) @ W3 + b3 + x
  k_gemm<0, true><<<dim3(DM / 64, MROWS / 128), 256, GEMM_LDS, stream>>>(
      yx2b, W3t, b3, x, out, MROWS, DM, PJ);
}
